// MHA_57655640981690
// MI455X (gfx1250) — matrix operand order repaired and checked by reading, not compiled, not run on hardware
//
#include <hip/hip_runtime.h>
#include <cstdint>

// ---------------------------------------------------------------------------
// MI455X (gfx1250, wave32) implementation of the reference transformer block.
//   - All GEMMs (A[M,512] · W[N,512]^T + bias) run on V_WMMA_F32_16X16X32_BF16
//     with f32 accumulation. Weights converted once to a bf16 cache; each GEMM
//     block stages a 64x256 bf16 tile in LDS via GLOBAL_LOAD_ASYNC_TO_LDS_B128
//     (ASYNCcnt-tracked, no VGPR round trip) when available, else batched b128
//     copies. Each wave computes a 32x64 output tile: 8 independent WMMAs per
//     32-K step off 4 shared B fragments.
//   - Attention: 1 wave / 16 query rows / (b,h); 16x1024 f32 score strip in
//     64KB LDS; QK^T and P·V both on WMMA.
//   - JAX threefry2x32 masks precomputed into bitmask words (2 outputs per
//     call -> half the threefry evaluations; wave32 ballot packs the bits).
// Workspace: ~100 MB (activations ping-pong + qkv + mask bits + bf16 weights).
// ---------------------------------------------------------------------------

#define NEGC  (-1.0e12f)
#define TOK   8192           // B * N = 8 * 1024
#define CDIM  512
#define GK    512            // shared K dimension of every GEMM

#if defined(__has_builtin)
#  if __has_builtin(__builtin_amdgcn_global_load_async_to_lds_b128)
#    define HAVE_ASYNC_COPY 1
#  endif
#endif
#ifndef HAVE_ASYNC_COPY
#  define HAVE_ASYNC_COPY 0
#endif

typedef __attribute__((ext_vector_type(16))) __bf16 v16bf;
typedef __attribute__((ext_vector_type(8)))  float  v8f;
typedef int v4ia __attribute__((vector_size(16)));   // builtin's int4 type

union BF16x16 { unsigned short u[16]; v16bf v; };
union BQ16    { uint4 q[2];           v16bf v; };
union US4     { unsigned short u[4];  uint2 d; };

__device__ __forceinline__ unsigned short f2bf(float f) {
  unsigned u = __builtin_bit_cast(unsigned, f);
  u += 0x7fffu + ((u >> 16) & 1u);          // round-to-nearest-even
  return (unsigned short)(u >> 16);
}

__device__ __forceinline__ v8f wmma_bf16(v16bf a, v16bf b, v8f c) {
  return __builtin_amdgcn_wmma_f32_16x16x32_bf16(false, a, false, b,
                                                 (short)0, c, false, false);
}

// A-fragment (16x32 bf16) from a row-major f32 row.  Contiguous K order (B fragments use the same):
// lane L: M = L&15, hv = L>>4;  elems 0..15 = row[k0 + hv*16 + 0..15]
__device__ __forceinline__ v16bf load_a_frag(const float* row, int k0, int hv,
                                             bool valid) {
  BF16x16 r;
  if (valid) {
    const float4 f0 = *(const float4*)(row + k0 + hv * 16);
    const float4 f1 = *(const float4*)(row + k0 + hv * 16 + 4);
    const float4 f2 = *(const float4*)(row + k0 + hv * 16 + 8);
    const float4 f3 = *(const float4*)(row + k0 + hv * 16 + 12);
    r.u[0] = f2bf(f0.x);  r.u[1] = f2bf(f0.y);  r.u[2]  = f2bf(f0.z);  r.u[3]  = f2bf(f0.w);
    r.u[4] = f2bf(f1.x);  r.u[5] = f2bf(f1.y);  r.u[6]  = f2bf(f1.z);  r.u[7]  = f2bf(f1.w);
    r.u[8] = f2bf(f2.x);  r.u[9] = f2bf(f2.y);  r.u[10] = f2bf(f2.z);  r.u[11] = f2bf(f2.w);
    r.u[12] = f2bf(f3.x); r.u[13] = f2bf(f3.y); r.u[14] = f2bf(f3.z); r.u[15] = f2bf(f3.w);
  } else {
#pragma unroll
    for (int i = 0; i < 16; i++) r.u[i] = 0;
  }
  return r.v;
}

// B-fragment (32x16): lane L needs 16 contiguous bf16 (K-run) of row N=L&15.
__device__ __forceinline__ v16bf load_b_frag_lds(const unsigned short* p) {
  BQ16 r;
  r.q[0] = *(const uint4*)p;
  r.q[1] = *(const uint4*)(p + 8);
  return r.v;
}
__device__ __forceinline__ v16bf load_b_frag_g(const float* p) {
  BF16x16 r;
#pragma unroll
  for (int i = 0; i < 16; i++) r.u[i] = f2bf(p[i]);
  return r.v;
}

// --------------------------- JAX threefry2x32 ------------------------------
__host__ __device__ inline void threefry2x32(unsigned k0, unsigned k1,
                                             unsigned x0, unsigned x1,
                                             unsigned* o0, unsigned* o1) {
  const unsigned ks0 = k0, ks1 = k1, ks2 = k0 ^ k1 ^ 0x1BD11BDAu;
  x0 += ks0; x1 += ks1;
#define TF_R4(a, b, c, d)                                   \
  x0 += x1; x1 = (x1 << (a)) | (x1 >> (32 - (a))); x1 ^= x0; \
  x0 += x1; x1 = (x1 << (b)) | (x1 >> (32 - (b))); x1 ^= x0; \
  x0 += x1; x1 = (x1 << (c)) | (x1 >> (32 - (c))); x1 ^= x0; \
  x0 += x1; x1 = (x1 << (d)) | (x1 >> (32 - (d))); x1 ^= x0;
  TF_R4(13, 15, 26, 6)  x0 += ks1; x1 += ks2 + 1u;
  TF_R4(17, 29, 16, 24) x0 += ks2; x1 += ks0 + 2u;
  TF_R4(13, 15, 26, 6)  x0 += ks0; x1 += ks1 + 3u;
  TF_R4(17, 29, 16, 24) x0 += ks1; x1 += ks2 + 4u;
  TF_R4(13, 15, 26, 6)  x0 += ks2; x1 += ks0 + 5u;
#undef TF_R4
  *o0 = x0; *o1 = x1;
}

__device__ __forceinline__ bool bern_bit(unsigned bits, float p) {
  const unsigned fb = (bits >> 9) | 0x3f800000u;   // uniform in [0,1)
  return (__builtin_bit_cast(float, fb) - 1.0f) < p;
}

// One threefry per index pair (i, i+halfn) -> 2 mask bits; ballot packs words.
__global__ __launch_bounds__(256) void bern_mask_kernel(
    unsigned k0, unsigned k1, unsigned halfn, float p, unsigned* words) {
  const unsigned i = blockIdx.x * 256u + threadIdx.x;
  if (i >= halfn) return;                 // grids sized so no wave splits here
  unsigned y0, y1;
  threefry2x32(k0, k1, i, halfn + i, &y0, &y1);
  const unsigned w0 = __builtin_amdgcn_ballot_w32(bern_bit(y0, p));
  const unsigned w1 = __builtin_amdgcn_ballot_w32(bern_bit(y1, p));
  if ((threadIdx.x & 31u) == 0u) {
    words[i >> 5] = w0;
    words[(halfn + i) >> 5] = w1;
  }
}

// --------------------- weights f32 -> bf16 cache ---------------------------
__global__ __launch_bounds__(256) void cvt_bf16_kernel(
    const float* __restrict__ in, unsigned short* __restrict__ out, int n) {
  const int base = (blockIdx.x * 256 + threadIdx.x) * 4;
  if (base >= n) return;
  const float4 f = *(const float4*)(in + base);
  US4 r;
  r.u[0] = f2bf(f.x); r.u[1] = f2bf(f.y); r.u[2] = f2bf(f.z); r.u[3] = f2bf(f.w);
  *(uint2*)(out + base) = r.d;
}

// --------------------- x [8,512,1024] -> xt [8,1024,512] -------------------
__global__ __launch_bounds__(256) void transpose_x_kernel(
    const float* __restrict__ x, float* __restrict__ xt) {
  __shared__ float tile[32][33];
  const int b = blockIdx.z;
  const int nb = blockIdx.x * 32, cb = blockIdx.y * 32;
  const int tx = threadIdx.x, ty = threadIdx.y;   // 32 x 8
#pragma unroll
  for (int i = 0; i < 4; i++) {
    const int c = cb + ty + i * 8;
    tile[ty + i * 8][tx] = x[((size_t)b * 512 + c) * 1024 + nb + tx];
  }
  __syncthreads();
#pragma unroll
  for (int i = 0; i < 4; i++) {
    const int n = nb + ty + i * 8;
    xt[((size_t)b * 1024 + n) * 512 + cb + tx] = tile[tx][ty + i * 8];
  }
}

// ------------------- out = A[M,512]·W[N,512]^T + bias ----------------------
// fuse: 0 = none, 1 = gelu then +mask*NEG, 2 = +mask*NEG.
// W is pre-converted bf16. Block = 4 waves; block tile 128(M) x 64(N); each
// wave computes 32x64 (two 16-row A sub-tiles sharing 4 B fragments -> 8
// independent WMMAs per 32-K step). W tile staged in LDS in two 256-K stages
// (rows padded to 264 ushorts = 528B so row stride rotates banks).
__global__ __launch_bounds__(128) void gemm_wmma_kernel(
    const float* __restrict__ A, const unsigned short* __restrict__ W,
    const float* __restrict__ bias, float* __restrict__ out,
    int M, int N, int fuse, const unsigned* __restrict__ mask) {
  __shared__ unsigned short Wl[64][264];
  const int tid = threadIdx.x;
  const int n0 = blockIdx.x * 64, m0 = blockIdx.y * 128;
  const int wave = tid >> 5, lane = tid & 31;
  const int ncol = lane & 15, hv = lane >> 4;
  const int mA = m0 + wave * 32 + ncol;
  const bool v0 = (mA < M), v1 = (mA + 16 < M);
  const float* arow0 = A + (size_t)mA * GK;
  const float* arow1 = arow0 + (size_t)16 * GK;
  v8f acc[2][4] = {};
  for (int kb = 0; kb < GK; kb += 256) {
    __syncthreads();
    // ---- stage 64 rows x 256 K of bf16 W into LDS -------------------------
#if HAVE_ASYNC_COPY
    // CDNA5 async copy: global -> LDS directly, tracked by ASYNCcnt.
    // Builtin params (per probe): (int4* global_src, int4* lds_dst, imm, imm);
    // HIP maps the builtin address spaces to cuda_device/cuda_shared which
    // accept generic pointers.
#pragma unroll
    for (int jj = 0; jj < 16; jj++) {
      const int i = tid + jj * 128;                      // 2048 b128 chunks
      const int row = i >> 5, ch = i & 31;
      __builtin_amdgcn_global_load_async_to_lds_b128(
          (v4ia*)(W + (size_t)(n0 + row) * GK + kb + ch * 8),
          (v4ia*)&Wl[row][ch * 8], 0, 0);
    }
#  if defined(__has_builtin) && __has_builtin(__builtin_amdgcn_s_wait_asynccnt)
    __builtin_amdgcn_s_wait_asynccnt(0);
#  else
    asm volatile("s_wait_asynccnt 0x0" ::: "memory");
#  endif
#else
    // Fallback: batched b128 copies, 4 loads in flight per step.
#pragma unroll
    for (int jj = 0; jj < 16; jj += 4) {
      uint4 t[4];
#pragma unroll
      for (int u = 0; u < 4; u++) {
        const int i = tid + (jj + u) * 128;
        const int row = i >> 5, ch = i & 31;
        t[u] = *(const uint4*)(W + (size_t)(n0 + row) * GK + kb + ch * 8);
      }
#pragma unroll
      for (int u = 0; u < 4; u++) {
        const int i = tid + (jj + u) * 128;
        const int row = i >> 5, ch = i & 31;
        *(uint4*)&Wl[row][ch * 8] = t[u];
      }
    }
#endif
    __syncthreads();
    if (kb == 0) {
      // prefetch second 256-K stage (64 rows x 512B) while stage 1 computes
#pragma unroll
      for (int u = 0; u < 2; u++) {
        const int i = tid + u * 128;                     // 256 cachelines
        const int row = i >> 2, seg = i & 3;
        __builtin_prefetch(W + (size_t)(n0 + row) * GK + 256 + seg * 64, 0, 3);
      }
    }
#pragma unroll
    for (int k0 = 0; k0 < 256; k0 += 32) {
      v16bf bfr[4];
#pragma unroll
      for (int t = 0; t < 4; t++)                        // issue all 8 ds b128
        bfr[t] = load_b_frag_lds(&Wl[t * 16 + ncol][k0 + hv * 16]);
      const v16bf a0 = load_a_frag(arow0, kb + k0, hv, v0);
      const v16bf a1 = load_a_frag(arow1, kb + k0, hv, v1);
#pragma unroll
      for (int t = 0; t < 4; t++) {                      // 8 independent WMMAs
        acc[0][t] = wmma_bf16(a0, bfr[t], acc[0][t]);
        acc[1][t] = wmma_bf16(a1, bfr[t], acc[1][t]);
      }
    }
  }
#pragma unroll
  for (int t = 0; t < 4; t++) {
    const int gn = n0 + t * 16 + ncol;
    const float bi = bias[gn];
#pragma unroll
    for (int s = 0; s < 2; s++) {
#pragma unroll
      for (int v = 0; v < 8; v++) {
        const int gm = m0 + wave * 32 + s * 16 + v + 8 * hv;  // C/D layout row
        if (gm >= M) continue;
        float val = acc[s][t][v] + bi;
        if (fuse == 1) val = 0.5f * val * (1.0f + erff(val * 0.70710678118f));
        if (fuse != 0) {
          const unsigned idx = (unsigned)gm * (unsigned)N + (unsigned)gn;
          if ((mask[idx >> 5] >> (idx & 31u)) & 1u) val += NEGC;
        }
        out[(size_t)gm * N + gn] = val;
      }
    }
  }
}

// ------------------------------ LayerNorm ----------------------------------
__global__ __launch_bounds__(256) void ln_kernel(
    const float* __restrict__ in, const float* __restrict__ g,
    const float* __restrict__ bt, float* __restrict__ out) {
  const int wave = threadIdx.x >> 5, lane = threadIdx.x & 31;
  const int tok = blockIdx.x * 8 + wave;
  const float* row = in + (size_t)tok * 512;
  float vals[16];
  float s = 0.f, s2 = 0.f;
#pragma unroll
  for (int i = 0; i < 16; i++) {
    const float x = row[lane + i * 32];
    vals[i] = x; s += x; s2 += x * x;
  }
#pragma unroll
  for (int d = 1; d < 32; d <<= 1) {
    s  += __shfl_xor(s, d, 32);
    s2 += __shfl_xor(s2, d, 32);
  }
  const float mu  = s * (1.0f / 512.0f);
  const float var = s2 * (1.0f / 512.0f) - mu * mu;
  const float rs  = rsqrtf(var + 1e-5f);
  float* orow = out + (size_t)tok * 512;
#pragma unroll
  for (int i = 0; i < 16; i++) {
    const int c = lane + i * 32;
    orow[c] = (vals[i] - mu) * rs * g[c] + bt[c];
  }
}

// ------------------------------ Attention ----------------------------------
// One wave handles 16 query rows of one (b,h). Scores 16x1024 f32 in LDS.
__global__ __launch_bounds__(32) void attn_wmma_kernel(
    const float* __restrict__ qkv, float* __restrict__ o,
    const unsigned* __restrict__ amask) {
  __shared__ float S[16 * 1024];                       // 64 KB of 320 KB/WGP
  const int bid = blockIdx.x;
  const int mt = bid & 63, h = (bid >> 6) & 7, b = bid >> 9;
  const int lane = threadIdx.x;
  const int ncol = lane & 15, hv = lane >> 4;
  const int mrow = mt * 16 + ncol;
  const float* qrow = qkv + ((size_t)(b * 1024 + mrow)) * 1536 + h * 64;
  const v16bf aq0 = load_a_frag(qrow, 0, hv, true);    // K 0..31
  const v16bf aq1 = load_a_frag(qrow, 32, hv, true);   // K 32..63
  const unsigned bhbase = ((unsigned)(b * 8 + h)) << 20;
  for (int j = 0; j < 64; j++) {                        // 16-col score tiles
    const int token = j * 16 + ncol;
    const float* krow = qkv + ((size_t)(b * 1024 + token)) * 1536 + 512 + h * 64;
    const v16bf bk0 = load_b_frag_g(krow + hv * 16);
    const v16bf bk1 = load_b_frag_g(krow + 32 + hv * 16);
    v8f acc = {};
    acc = wmma_bf16(aq0, bk0, acc);
    acc = wmma_bf16(aq1, bk1, acc);
#pragma unroll
    for (int v = 0; v < 8; v++) {
      const int srow = v + 8 * hv;
      const int scol = j * 16 + ncol;
      float sv = acc[v] * 0.125f;                      // 1/sqrt(64)
      const unsigned idx = bhbase + (unsigned)((mt * 16 + srow) * 1024 + scol);
      if ((amask[idx >> 5] >> (idx & 31u)) & 1u) sv += NEGC;
      S[srow * 1024 + scol] = sv;
    }
  }
  __syncthreads();
  {                                                    // softmax, 2 lanes/row
    float* row = S + (lane & 15) * 1024 + hv * 512;
    float mx = -3.0e38f;
    for (int c = 0; c < 512; c++) mx = fmaxf(mx, row[c]);
    mx = fmaxf(mx, __shfl_xor(mx, 16, 32));
    float sum = 0.f;
    for (int c = 0; c < 512; c++) { const float e = __expf(row[c] - mx); row[c] = e; sum += e; }
    sum += __shfl_xor(sum, 16, 32);
    const float inv = 1.0f / sum;
    for (int c = 0; c < 512; c++) row[c] *= inv;
  }
  __syncthreads();
  v8f oacc[4] = {};                                    // O = P(16x1024)·V
  const float* prow0 = S + (lane & 15) * 1024;
  for (int kk = 0; kk < 32; kk++) {
    const v16bf ap = load_a_frag(prow0 + kk * 32, 0, hv, true);
#pragma unroll
    for (int t = 0; t < 4; t++) {
      BF16x16 r;
#pragma unroll
      for (int e = 0; e < 16; e++) {
        const int token = kk * 32 + hv * 16 + e;
        r.u[e] = f2bf(qkv[((size_t)(b * 1024 + token)) * 1536 + 1024 + h * 64 +
                          t * 16 + ncol]);
      }
      oacc[t] = wmma_bf16(ap, r.v, oacc[t]);
    }
  }
#pragma unroll
  for (int t = 0; t < 4; t++) {
#pragma unroll
    for (int v = 0; v < 8; v++) {
      const int grow = mt * 16 + v + 8 * hv;
      const int gcol = h * 64 + t * 16 + ncol;
      o[((size_t)(b * 1024 + grow)) * 512 + gcol] = oacc[t][v];
    }
  }
}

// --------------------------- mean over tokens ------------------------------
__global__ __launch_bounds__(256) void pool_kernel(
    const float* __restrict__ in, float* __restrict__ out) {
  const int i = blockIdx.x * 256 + threadIdx.x;        // 4096 = 8*512
  const int b = i >> 9, c = i & 511;
  const float* p = in + ((size_t)b * 1024) * 512 + c;
  float s = 0.f;
  for (int n = 0; n < 1024; n++) s += p[(size_t)n * 512];
  out[i] = s * (1.0f / 1024.0f);
}

// ---------------------------------------------------------------------------
extern "C" void kernel_launch(void* const* d_in, const int* in_sizes, int n_in,
                              void* d_out, int out_size, void* d_ws, size_t ws_size,
                              hipStream_t stream) {
  (void)in_sizes; (void)n_in; (void)out_size; (void)ws_size;
  const float* x      = (const float*)d_in[0];
  const float* conv_w = (const float*)d_in[1];
  const float* conv_b = (const float*)d_in[2];
  const float* ln_g   = (const float*)d_in[3];
  const float* ln_b   = (const float*)d_in[4];
  const float* qkv_w  = (const float*)d_in[5];
  const float* qkv_b  = (const float*)d_in[6];
  const float* proj_w = (const float*)d_in[7];
  const float* proj_b = (const float*)d_in[8];
  const float* w1     = (const float*)d_in[9];
  const float* b1     = (const float*)d_in[10];
  const float* w2     = (const float*)d_in[11];
  const float* b2     = (const float*)d_in[12];
  const float* fc_w   = (const float*)d_in[13];
  const float* fc_b   = (const float*)d_in[14];
  float* out = (float*)d_out;

  // Workspace layout (~100 MB).
  float* buf0 = (float*)d_ws;                          // 8192x512  f32
  float* buf1 = buf0 + (size_t)TOK * CDIM;             // 8192x512  f32
  float* buf2 = buf1 + (size_t)TOK * CDIM;             // 8192x1536 f32
  unsigned* maskA = (unsigned*)(buf2 + (size_t)TOK * 1536);  // 67M bits
  unsigned* mask1 = maskA + (8u * 8u * 1024u * 1024u) / 32u; // 4.2M bits
  unsigned* mask2 = mask1 + ((unsigned)TOK * CDIM) / 32u;
  float* pooled   = (float*)(mask2 + ((unsigned)TOK * CDIM) / 32u); // 8x512
  unsigned short* wc_conv = (unsigned short*)(pooled + 8 * 512);
  unsigned short* wc_qkv  = wc_conv + (size_t)512 * 512;
  unsigned short* wc_proj = wc_qkv + (size_t)1536 * 512;
  unsigned short* wc_w1   = wc_proj + (size_t)512 * 512;
  unsigned short* wc_w2   = wc_w1 + (size_t)512 * 512;
  unsigned short* wc_fc   = wc_w2 + (size_t)512 * 512;

  // Host-side: mk = jax.random.split(key(42), 3) via threefry2x32.
  unsigned y00, y01, y10, y11, y20, y21;
  threefry2x32(0u, 42u, 0u, 3u, &y00, &y01);
  threefry2x32(0u, 42u, 1u, 4u, &y10, &y11);
  threefry2x32(0u, 42u, 2u, 5u, &y20, &y21);
  // bits = [y00,y10,y20,y01,y11,y21] reshaped (3,2)
  const unsigned mk0a = y00, mk0b = y10;
  const unsigned mk1a = y20, mk1b = y01;
  const unsigned mk2a = y11, mk2b = y21;

  const unsigned halfA = 8u * 8u * 1024u * 1024u / 2u;   // 33554432
  const unsigned halfM = (unsigned)TOK * CDIM / 2u;      // 2097152
  bern_mask_kernel<<<halfA / 256, 256, 0, stream>>>(mk0a, mk0b, halfA, 0.5f, maskA);
  bern_mask_kernel<<<halfM / 256, 256, 0, stream>>>(mk1a, mk1b, halfM, 0.3f, mask1);
  bern_mask_kernel<<<halfM / 256, 256, 0, stream>>>(mk2a, mk2b, halfM, 0.1f, mask2);

  // bf16 weight cache (one conversion pass; GEMMs then stream bf16 from L2)
  cvt_bf16_kernel<<<256, 256, 0, stream>>>(conv_w, wc_conv, 512 * 512);
  cvt_bf16_kernel<<<768, 256, 0, stream>>>(qkv_w, wc_qkv, 1536 * 512);
  cvt_bf16_kernel<<<256, 256, 0, stream>>>(proj_w, wc_proj, 512 * 512);
  cvt_bf16_kernel<<<256, 256, 0, stream>>>(w1, wc_w1, 512 * 512);
  cvt_bf16_kernel<<<256, 256, 0, stream>>>(w2, wc_w2, 512 * 512);
  cvt_bf16_kernel<<<1280, 256, 0, stream>>>(fc_w, wc_fc, 2560 * 512);

  transpose_x_kernel<<<dim3(32, 16, 8), dim3(32, 8), 0, stream>>>(x, buf0);
  // conv1x1 (per-pixel linear)
  gemm_wmma_kernel<<<dim3(8, 64), 128, 0, stream>>>(buf0, wc_conv, conv_b, buf1, TOK, 512, 0, nullptr);
  ln_kernel<<<1024, 256, 0, stream>>>(buf1, ln_g, ln_b, buf0);
  gemm_wmma_kernel<<<dim3(24, 64), 128, 0, stream>>>(buf0, wc_qkv, qkv_b, buf2, TOK, 1536, 0, nullptr);
  attn_wmma_kernel<<<4096, 32, 0, stream>>>(buf2, buf1, maskA);
  gemm_wmma_kernel<<<dim3(8, 64), 128, 0, stream>>>(buf1, wc_proj, proj_b, buf0, TOK, 512, 0, nullptr);
  gemm_wmma_kernel<<<dim3(8, 64), 128, 0, stream>>>(buf0, wc_w1, b1, buf1, TOK, 512, 1, mask1);
  gemm_wmma_kernel<<<dim3(8, 64), 128, 0, stream>>>(buf1, wc_w2, b2, buf0, TOK, 512, 2, mask2);
  ln_kernel<<<1024, 256, 0, stream>>>(buf0, ln_g, ln_b, buf1);
  pool_kernel<<<16, 256, 0, stream>>>(buf1, pooled);
  // final fc: M=8 padded into one 128-row tile (guarded loads/stores)
  gemm_wmma_kernel<<<dim3(40, 1), 128, 0, stream>>>(pooled, wc_fc, fc_b, out, 8, 2560, 0, nullptr);
}
